// model_4v1_27736898798378
// MI455X (gfx1250) — matrix operand order repaired and checked by reading, not compiled, not run on hardware
//
#include <hip/hip_runtime.h>
#include <hip/hip_bf16.h>
#include <float.h>

// Problem constants (match reference)
#define NND   100000   // nodes
#define DD    128      // feature dim
#define HIDD  256      // hidden dim
#define MDEG  16       // max degree
#define NBLK  5        // residual blocks
#define NGR   2048     // graphs
#define CATD  768      // D*(BLOCKS+1)
#define ROWT  6250     // NND/16 row tiles (exact)
#define GROWT 128      // NGR/16 row tiles (exact)

typedef __attribute__((ext_vector_type(16))) __bf16 v16bf;
typedef __attribute__((ext_vector_type(8)))  float  v8f;

static __device__ __forceinline__ unsigned short f2bf(float f) {
  unsigned int u = __float_as_uint(f);
  u += 0x7FFFu + ((u >> 16) & 1u);           // round-to-nearest-even
  return (unsigned short)(u >> 16);
}

// ---------------------------------------------------------------------------
// Weight packing: W (Kin x Kout, f32 row-major) -> bf16 B-fragment layout.
// Per 32x16 (KxN) tile: 32 lanes x 16 bf16 contiguous. Lane holds column
// n = nt*16 + (lane&15), K block selected by lane>>4, element e = k%16.
// ---------------------------------------------------------------------------
__global__ void pack_w(const float* __restrict__ W, unsigned short* __restrict__ P,
                       int Kin, int Kout) {
  int total = Kin * Kout;
  int NT = Kout >> 4;
  for (int i = blockIdx.x * blockDim.x + threadIdx.x; i < total;
       i += gridDim.x * blockDim.x) {
    int e    = i & 15;
    int lane = (i >> 4) & 31;
    int tile = i >> 9;
    int nt   = tile % NT;
    int kt   = tile / NT;
    int k = kt * 32 + ((lane >> 4) & 1) * 16 + e;
    int n = nt * 16 + (lane & 15);
    P[i] = f2bf(W[(size_t)k * Kout + n]);
  }
}

// ---------------------------------------------------------------------------
// bf16 WMMA GEMM: out(rows x Kout) = A(rows x Kin, bf16) @ Bpacked + bias
// Each wave: one 16-row tile x 64 cols (4 fragments). Optional residual
// (out += (1+eps[idx]) * resid), optional relu, writes f32 and/or bf16.
// ---------------------------------------------------------------------------
__global__ void wmma_gemm_bf16(const unsigned short* __restrict__ A,
                               const unsigned short* __restrict__ Bp,
                               const float* __restrict__ bias,
                               const float* __restrict__ resid,
                               const float* __restrict__ eps_ptr, int eps_idx,
                               float* __restrict__ outf,
                               unsigned short* __restrict__ outbf,
                               int row_tiles, int Kin, int Kout,
                               int lda, int ldo, int do_relu) {
  const int wave = threadIdx.x >> 5;
  const int lane = threadIdx.x & 31;
  const int rt = blockIdx.x * (blockDim.x >> 5) + wave;
  if (rt >= row_tiles) return;                // wave-uniform: EXEC all-1 inside
  const int colb = blockIdx.y * 64;
  const int NT = Kout >> 4;
  const int m = lane & 15;
  const int half = (lane >> 4) & 1;

  union FR { v16bf v; uint4 u[2]; };
  v8f acc0 = {}, acc1 = {}, acc2 = {}, acc3 = {};

  const unsigned short* arow = A + (size_t)(rt * 16 + m) * lda + half * 16;
  const int ktiles = Kin >> 5;
  for (int kt = 0; kt < ktiles; ++kt) {
    FR fa;
    const unsigned short* ap = arow + kt * 32;
    fa.u[0] = *(const uint4*)(ap);        // K = kt*32 + half*16 + 0..7
    fa.u[1] = *(const uint4*)(ap + 8);    // K = kt*32 + half*16 + 8..15
    const unsigned short* bp =
        Bp + ((size_t)(kt * NT + (colb >> 4)) * 32 + lane) * 16;
    FR fb0, fb1, fb2, fb3;
    fb0.u[0] = *(const uint4*)(bp + 0 * 512);
    fb0.u[1] = *(const uint4*)(bp + 0 * 512 + 8);
    fb1.u[0] = *(const uint4*)(bp + 1 * 512);
    fb1.u[1] = *(const uint4*)(bp + 1 * 512 + 8);
    fb2.u[0] = *(const uint4*)(bp + 2 * 512);
    fb2.u[1] = *(const uint4*)(bp + 2 * 512 + 8);
    fb3.u[0] = *(const uint4*)(bp + 3 * 512);
    fb3.u[1] = *(const uint4*)(bp + 3 * 512 + 8);
    acc0 = __builtin_amdgcn_wmma_f32_16x16x32_bf16(false, fa.v, false, fb0.v,
                                                   (short)0, acc0, false, false);
    acc1 = __builtin_amdgcn_wmma_f32_16x16x32_bf16(false, fa.v, false, fb1.v,
                                                   (short)0, acc1, false, false);
    acc2 = __builtin_amdgcn_wmma_f32_16x16x32_bf16(false, fa.v, false, fb2.v,
                                                   (short)0, acc2, false, false);
    acc3 = __builtin_amdgcn_wmma_f32_16x16x32_bf16(false, fa.v, false, fb3.v,
                                                   (short)0, acc3, false, false);
  }

  float scale = 1.0f;
  if (resid && eps_ptr) scale = 1.0f + eps_ptr[eps_idx];
  const int rbase = rt * 16 + half * 8;     // C layout: lanes 16-31 hold M+8
  v8f accs[4] = {acc0, acc1, acc2, acc3};
#pragma unroll
  for (int j = 0; j < 4; ++j) {
    int col = colb + j * 16 + (lane & 15);
    float bv = bias ? bias[col] : 0.0f;
#pragma unroll
    for (int v = 0; v < 8; ++v) {
      int row = rbase + v;
      float f = accs[j][v] + bv;
      if (resid) f += scale * resid[(size_t)row * ldo + col];
      if (do_relu) f = fmaxf(f, 0.0f);
      size_t o = (size_t)row * ldo + col;
      if (outbf) outbf[o] = f2bf(f);
      if (outf)  outf[o]  = f;
    }
  }
}

// ---------------------------------------------------------------------------
// Elementwise / reduction helper kernels
// ---------------------------------------------------------------------------
__global__ void copy_f32(const float* __restrict__ src, float* __restrict__ dst,
                         int count) {
  for (int i = blockIdx.x * blockDim.x + threadIdx.x; i < count;
       i += gridDim.x * blockDim.x) dst[i] = src[i];
}

__global__ void cvt_f32_bf16(const float* __restrict__ src,
                             unsigned short* __restrict__ dst, int count) {
  for (int i = blockIdx.x * blockDim.x + threadIdx.x; i < count;
       i += gridDim.x * blockDim.x) dst[i] = f2bf(src[i]);
}

// Per-column min over n rows (stage A: 256 partials; stage B: finalize)
__global__ void colmin_partial(const float* __restrict__ h, float* __restrict__ part,
                               int n) {
  int d = threadIdx.x;
  float m = FLT_MAX;
  for (int r = blockIdx.x; r < n; r += gridDim.x)
    m = fminf(m, h[(size_t)r * DD + d]);
  part[blockIdx.x * DD + d] = m;
}
__global__ void colmin_final(const float* __restrict__ part, float* __restrict__ cmin,
                             int nparts) {
  int d = threadIdx.x;
  float m = FLT_MAX;
  for (int b = 0; b < nparts; ++b) m = fminf(m, part[b * DD + d]);
  cmin[d] = m;
}

// Neighbor aggregation: sum + max (dummy row -> 0 for sum, colmin for max);
// writes merge-GEMM input directly as bf16: cols [0,128)=sum, [128,256)=max.
__global__ void aggregate(const float* __restrict__ h, const int* __restrict__ nbrs,
                          const float* __restrict__ cmin,
                          unsigned short* __restrict__ outA) {
  __shared__ int nb[MDEG];
  int n = blockIdx.x, d = threadIdx.x;
  if (d < MDEG) nb[d] = nbrs[(size_t)n * MDEG + d];
  __syncthreads();
  float cm = cmin[d];
  float s = 0.0f, mx = -FLT_MAX;
#pragma unroll
  for (int j = 0; j < MDEG; ++j) {
    int t = nb[j];
    if (t >= 0) {
      float v = h[(size_t)t * DD + d];
      s += v;
      mx = fmaxf(mx, v);
    } else {
      mx = fmaxf(mx, cm);
    }
  }
  outA[(size_t)n * HIDD + d]      = f2bf(s);
  outA[(size_t)n * HIDD + DD + d] = f2bf(mx);
}

// BatchNorm stats (biased variance), two stage; then affine fold a*x + c.
__global__ void bn_partial(const float* __restrict__ z, float* __restrict__ psum,
                           float* __restrict__ psq, int n) {
  int d = threadIdx.x;
  float s = 0.0f, q = 0.0f;
  for (int r = blockIdx.x; r < n; r += gridDim.x) {
    float v = z[(size_t)r * DD + d];
    s += v; q += v * v;
  }
  psum[blockIdx.x * DD + d] = s;
  psq[blockIdx.x * DD + d]  = q;
}
__global__ void bn_final(const float* __restrict__ psum, const float* __restrict__ psq,
                         const float* __restrict__ gg, const float* __restrict__ bb,
                         int n, int nparts, float* __restrict__ a,
                         float* __restrict__ c) {
  int d = threadIdx.x;
  float s = 0.0f, q = 0.0f;
  for (int b = 0; b < nparts; ++b) { s += psum[b * DD + d]; q += psq[b * DD + d]; }
  float mean = s / (float)n;
  float var  = q / (float)n - mean * mean;
  float av = gg[d] * rsqrtf(var + 1e-5f);
  a[d] = av;
  c[d] = bb[d] - av * mean;
}
__global__ void bn_apply(float* __restrict__ z, const float* __restrict__ a,
                         const float* __restrict__ c, float* __restrict__ h,
                         int count) {
  for (int i = blockIdx.x * blockDim.x + threadIdx.x; i < count;
       i += gridDim.x * blockDim.x) {
    int d = i & (DD - 1);
    float v = a[d] * z[i] + c[d];
    z[i] = v;
    if (h) h[i] += v;
  }
}

// Segment-sum a 128-col block into g[:, colofs..colofs+128) (membership sorted).
__global__ void segsum(const float* __restrict__ src, const int* __restrict__ mem,
                       int n, float* __restrict__ g, int colofs) {
  int gid = blockIdx.x, d = threadIdx.x;
  // lower_bound(mem, gid) and lower_bound(mem, gid+1)
  int lo = 0, hi = n;
  while (lo < hi) { int mid = (lo + hi) >> 1; if (mem[mid] < gid) lo = mid + 1; else hi = mid; }
  int lo2 = lo, hi2 = n;
  while (lo2 < hi2) { int mid = (lo2 + hi2) >> 1; if (mem[mid] < gid + 1) lo2 = mid + 1; else hi2 = mid; }
  float s = 0.0f;
  for (int r = lo; r < lo2; ++r) s += src[(size_t)r * DD + d];
  g[(size_t)gid * CATD + colofs + d] = s;
}

// Final projection: (NGR x 128) @ (128 x 1) + b
__global__ void dense2(const float* __restrict__ t, const float* __restrict__ w,
                       const float* __restrict__ b, float* __restrict__ out, int n) {
  int r = blockIdx.x * blockDim.x + threadIdx.x;
  if (r >= n) return;
  float acc = b[0];
#pragma unroll
  for (int d = 0; d < DD; ++d) acc += t[(size_t)r * DD + d] * w[d];
  out[r] = acc;
}

// ---------------------------------------------------------------------------
// Host-side pipeline
// ---------------------------------------------------------------------------
extern "C" void kernel_launch(void* const* d_in, const int* in_sizes, int n_in,
                              void* d_out, int out_size, void* d_ws, size_t ws_size,
                              hipStream_t stream) {
  const float* x       = (const float*)d_in[0];
  const int*   nbrs    = (const int*)d_in[1];
  const int*   mem     = (const int*)d_in[2];
  const float* merge_W = (const float*)d_in[3];
  const float* merge_b = (const float*)d_in[4];
  const float* lin1_W  = (const float*)d_in[5];
  const float* lin1_b  = (const float*)d_in[6];
  const float* lin2_W  = (const float*)d_in[7];
  const float* lin2_b  = (const float*)d_in[8];
  const float* bn_g    = (const float*)d_in[9];
  const float* bn_b    = (const float*)d_in[10];
  const float* eps     = (const float*)d_in[11];
  const float* d0W     = (const float*)d_in[12];
  const float* d0b     = (const float*)d_in[13];
  const float* d1W     = (const float*)d_in[14];
  const float* d1b     = (const float*)d_in[15];
  const float* bn0g    = (const float*)d_in[16];
  const float* bn0b    = (const float*)d_in[17];
  const float* d2W     = (const float*)d_in[18];
  const float* d2b     = (const float*)d_in[19];
  float* out = (float*)d_out;

  char* base = (char*)d_ws;
  size_t off = 0;
  auto take = [&](size_t bytes) -> void* {
    void* p = base + off;
    off = (off + bytes + 255) & ~(size_t)255;
    return p;
  };
  float* h            = (float*)take((size_t)NND * DD * 4);
  float* zf           = (float*)take((size_t)NND * DD * 4);
  unsigned short* bfA = (unsigned short*)take((size_t)NND * HIDD * 2);
  unsigned short* bfB = (unsigned short*)take((size_t)NND * DD * 2);
  float* cmin         = (float*)take(DD * 4);
  float* pmin         = (float*)take(256 * DD * 4);
  float* psum         = (float*)take(256 * DD * 4);
  float* psq          = (float*)take(256 * DD * 4);
  float* bna          = (float*)take(DD * 4);
  float* bnc          = (float*)take(DD * 4);
  float* g            = (float*)take((size_t)NGR * CATD * 4);
  unsigned short* gbf = (unsigned short*)take((size_t)NGR * CATD * 2);
  unsigned short* t0bf= (unsigned short*)take((size_t)NGR * HIDD * 2);
  float* t1f          = (float*)take((size_t)NGR * DD * 4);
  unsigned short* pmg = (unsigned short*)take((size_t)NBLK * HIDD * DD * 2);
  unsigned short* pl1 = (unsigned short*)take((size_t)NBLK * DD * HIDD * 2);
  unsigned short* pl2 = (unsigned short*)take((size_t)NBLK * HIDD * DD * 2);
  unsigned short* pd0 = (unsigned short*)take((size_t)CATD * HIDD * 2);
  unsigned short* pd1 = (unsigned short*)take((size_t)HIDD * DD * 2);

  // Pack all weights into bf16 B-fragment layout (tiny, once per call)
  for (int i = 0; i < NBLK; ++i) {
    pack_w<<<(HIDD * DD + 255) / 256, 256, 0, stream>>>(
        merge_W + (size_t)i * HIDD * DD, pmg + (size_t)i * HIDD * DD, HIDD, DD);
    pack_w<<<(DD * HIDD + 255) / 256, 256, 0, stream>>>(
        lin1_W + (size_t)i * DD * HIDD, pl1 + (size_t)i * DD * HIDD, DD, HIDD);
    pack_w<<<(HIDD * DD + 255) / 256, 256, 0, stream>>>(
        lin2_W + (size_t)i * HIDD * DD, pl2 + (size_t)i * HIDD * DD, HIDD, DD);
  }
  pack_w<<<(CATD * HIDD + 255) / 256, 256, 0, stream>>>(d0W, pd0, CATD, HIDD);
  pack_w<<<(HIDD * DD + 255) / 256, 256, 0, stream>>>(d1W, pd1, HIDD, DD);

  // h = x ; readout of x into g[:, 0:128)
  copy_f32<<<2048, 256, 0, stream>>>(x, h, NND * DD);
  segsum<<<NGR, DD, 0, stream>>>(x, mem, NND, g, 0);

  const dim3 blk(128);                     // 4 waves/block
  const dim3 gN128((ROWT + 3) / 4, 2);     // Kout=128
  const dim3 gN256((ROWT + 3) / 4, 4);     // Kout=256

  for (int i = 0; i < NBLK; ++i) {
    colmin_partial<<<256, DD, 0, stream>>>(h, pmin, NND);
    colmin_final<<<1, DD, 0, stream>>>(pmin, cmin, 256);
    aggregate<<<NND, DD, 0, stream>>>(h, nbrs, cmin, bfA);
    // merge: (N x 256)@(256 x 128) + b + (1+eps_i)*h  -> bf16
    wmma_gemm_bf16<<<gN128, blk, 0, stream>>>(
        bfA, pmg + (size_t)i * HIDD * DD, merge_b + i * DD, h, eps, i,
        nullptr, bfB, ROWT, HIDD, DD, HIDD, DD, 0);
    // lin1: relu((N x 128)@(128 x 256) + b) -> bf16
    wmma_gemm_bf16<<<gN256, blk, 0, stream>>>(
        bfB, pl1 + (size_t)i * DD * HIDD, lin1_b + i * HIDD, nullptr, nullptr, 0,
        nullptr, bfA, ROWT, DD, HIDD, DD, HIDD, 1);
    // lin2: relu((N x 256)@(256 x 128) + b) -> f32
    wmma_gemm_bf16<<<gN128, blk, 0, stream>>>(
        bfA, pl2 + (size_t)i * HIDD * DD, lin2_b + i * DD, nullptr, nullptr, 0,
        zf, nullptr, ROWT, HIDD, DD, HIDD, DD, 1);
    // BatchNorm over N rows, then z->cat column block + dense connection h += z
    bn_partial<<<256, DD, 0, stream>>>(zf, psum, psq, NND);
    bn_final<<<1, DD, 0, stream>>>(psum, psq, bn_g + i * DD, bn_b + i * DD,
                                   NND, 256, bna, bnc);
    bn_apply<<<2048, 256, 0, stream>>>(zf, bna, bnc, h, NND * DD);
    segsum<<<NGR, DD, 0, stream>>>(zf, mem, NND, g, (i + 1) * DD);
  }

  // Graph head
  cvt_f32_bf16<<<(NGR * CATD + 255) / 256, 256, 0, stream>>>(g, gbf, NGR * CATD);
  wmma_gemm_bf16<<<dim3((GROWT + 3) / 4, 4), blk, 0, stream>>>(
      gbf, pd0, d0b, nullptr, nullptr, 0, nullptr, t0bf,
      GROWT, CATD, HIDD, CATD, HIDD, 1);
  wmma_gemm_bf16<<<dim3((GROWT + 3) / 4, 2), blk, 0, stream>>>(
      t0bf, pd1, d1b, nullptr, nullptr, 0, t1f, nullptr,
      GROWT, HIDD, DD, HIDD, DD, 1);
  bn_partial<<<256, DD, 0, stream>>>(t1f, psum, psq, NGR);
  bn_final<<<1, DD, 0, stream>>>(psum, psq, bn0g, bn0b, NGR, 256, bna, bnc);
  bn_apply<<<(NGR * DD + 255) / 256, 256, 0, stream>>>(t1f, bna, bnc, nullptr,
                                                       NGR * DD);
  dense2<<<(NGR + 255) / 256, 256, 0, stream>>>(t1f, d2W, d2b, out, NGR);
}
